// BERTStudentPruner_20753281974878
// MI455X (gfx1250) — matrix operand order repaired and checked by reading, not compiled, not run on hardware
//
#include <hip/hip_runtime.h>
#include <hip/hip_bf16.h>

// ---------------------------------------------------------------------------
// BERT student pruner: y_soft = sigmoid(MLP3(x)), y_hard = (rank(y_soft) < k)
// B=64, S=2048, D=768, k=1024.
// bf16 WMMA (v_wmma_f32_16x16x32_bf16) fused 2-layer GEMM, intermediates in
// LDS, weights bf16-converted once (L2-resident). Each wave: 2 M-tiles x
// 4 N-tiles per group -> 8 WMMAs per 8 global_load_b128 (1:1).
// ---------------------------------------------------------------------------

typedef __attribute__((ext_vector_type(16))) __bf16 v16bf;
typedef __attribute__((ext_vector_type(8)))  float  v8f;

#define DH        768
#define TOK_TILE  64
#define LDS_PITCH 776     // 768 + 8 bf16 pad: row stride 1552B = 388 dwords -> conflict-free b128
#define SEQ       2048

struct alignas(16) U128 { unsigned int x, y, z, w; };

union FragU {
  U128  u[2];
  v16bf v;
};

static __device__ __forceinline__ unsigned short f2bf(float f) {
  unsigned int u = __float_as_uint(f);
  u += 0x7FFFu + ((u >> 16) & 1u);        // round-to-nearest-even
  return (unsigned short)(u >> 16);
}
static __device__ __forceinline__ float bf2f(unsigned short h) {
  return __uint_as_float(((unsigned int)h) << 16);
}

// One dense layer for this wave's strip:
//   out[mg..mg+31][nq..nq+191] = relu(in . W^T + bias)
// bf16 in/out via LDS, bf16 weights streamed from L2.
static __device__ __forceinline__ void mlp_layer(
    const unsigned short* __restrict__ sIn,   // [TOK_TILE][LDS_PITCH] bf16
    unsigned short*       __restrict__ sOut,  // [TOK_TILE][LDS_PITCH] bf16
    const unsigned short* __restrict__ W,     // [DH][DH] bf16, n-major / k-contiguous
    const float*          __restrict__ bias,  // [DH]
    int mg, int nq, int lane15, int laneHalf)
{
  // A fragment: row m = mg(+16) + lane15; lane holds K = klo..klo+15 (klo = 16*laneHalf)
  const unsigned short* aBase0 = sIn + (mg + lane15) * LDS_PITCH + 16 * laneHalf;
  const unsigned short* aBase1 = aBase0 + 16 * LDS_PITCH;

  for (int ng = 0; ng < 3; ++ng) {          // 3 groups x 4 N-tiles = 12 tiles (192 cols)
    const int nb = nq + ng * 64;
    v8f acc0[4] = {v8f{}, v8f{}, v8f{}, v8f{}};   // M-tile 0
    v8f acc1[4] = {v8f{}, v8f{}, v8f{}, v8f{}};   // M-tile 1

    // B fragment: lane holds 16 contiguous k-values of W row n = ntile + lane15
    const unsigned short* wp0 = W + (size_t)(nb +  0 + lane15) * DH + 16 * laneHalf;
    const unsigned short* wp1 = W + (size_t)(nb + 16 + lane15) * DH + 16 * laneHalf;
    const unsigned short* wp2 = W + (size_t)(nb + 32 + lane15) * DH + 16 * laneHalf;
    const unsigned short* wp3 = W + (size_t)(nb + 48 + lane15) * DH + 16 * laneHalf;

#pragma unroll 2
    for (int kt = 0; kt < DH / 32; ++kt) {  // 24 k-steps of 32
      const int k0 = kt * 32;
      FragU a0, a1;
      a0.u[0] = *(const U128*)(aBase0 + k0);
      a0.u[1] = *(const U128*)(aBase0 + k0 + 8);
      a1.u[0] = *(const U128*)(aBase1 + k0);
      a1.u[1] = *(const U128*)(aBase1 + k0 + 8);

      FragU b0, b1, b2, b3;
      b0.u[0] = *(const U128*)(wp0 + k0); b0.u[1] = *(const U128*)(wp0 + k0 + 8);
      b1.u[0] = *(const U128*)(wp1 + k0); b1.u[1] = *(const U128*)(wp1 + k0 + 8);
      b2.u[0] = *(const U128*)(wp2 + k0); b2.u[1] = *(const U128*)(wp2 + k0 + 8);
      b3.u[0] = *(const U128*)(wp3 + k0); b3.u[1] = *(const U128*)(wp3 + k0 + 8);

      acc0[0] = __builtin_amdgcn_wmma_f32_16x16x32_bf16(false, a0.v, false, b0.v, (short)0, acc0[0], false, false);
      acc1[0] = __builtin_amdgcn_wmma_f32_16x16x32_bf16(false, a1.v, false, b0.v, (short)0, acc1[0], false, false);
      acc0[1] = __builtin_amdgcn_wmma_f32_16x16x32_bf16(false, a0.v, false, b1.v, (short)0, acc0[1], false, false);
      acc1[1] = __builtin_amdgcn_wmma_f32_16x16x32_bf16(false, a1.v, false, b1.v, (short)0, acc1[1], false, false);
      acc0[2] = __builtin_amdgcn_wmma_f32_16x16x32_bf16(false, a0.v, false, b2.v, (short)0, acc0[2], false, false);
      acc1[2] = __builtin_amdgcn_wmma_f32_16x16x32_bf16(false, a1.v, false, b2.v, (short)0, acc1[2], false, false);
      acc0[3] = __builtin_amdgcn_wmma_f32_16x16x32_bf16(false, a0.v, false, b3.v, (short)0, acc0[3], false, false);
      acc1[3] = __builtin_amdgcn_wmma_f32_16x16x32_bf16(false, a1.v, false, b3.v, (short)0, acc1[3], false, false);
    }

    // Epilogue: bias + relu + bf16 store. D layout: VGPR r -> row m0+r+8*laneHalf,
    // col = ntile + lane15.
#pragma unroll
    for (int j = 0; j < 4; ++j) {
      const int col = nb + j * 16 + lane15;
      const float bv = bias[col];
#pragma unroll
      for (int r = 0; r < 8; ++r) {
        float x0 = fmaxf(acc0[j][r] + bv, 0.0f);
        float x1 = fmaxf(acc1[j][r] + bv, 0.0f);
        const int rr = r + 8 * laneHalf;
        sOut[(mg + rr) * LDS_PITCH + col]      = f2bf(x0);
        sOut[(mg + 16 + rr) * LDS_PITCH + col] = f2bf(x1);
      }
    }
  }
}

// ---------------------------------------------------------------------------
__global__ void cvt_w_kernel(const float* __restrict__ W1,
                             const float* __restrict__ W2,
                             unsigned short* __restrict__ out) {
  const int N = DH * DH;
  int i = blockIdx.x * blockDim.x + threadIdx.x;
  if (i < N) {
    out[i]     = f2bf(W1[i]);
    out[N + i] = f2bf(W2[i]);
  }
}

// ---------------------------------------------------------------------------
__global__ void __launch_bounds__(256, 1)
mlp_score_kernel(const float* __restrict__ X,
                 const unsigned short* __restrict__ W1bf, const float* __restrict__ b1,
                 const unsigned short* __restrict__ W2bf, const float* __restrict__ b2,
                 const float* __restrict__ W3,  const float* __restrict__ b3,
                 float* __restrict__ ySoft)
{
  extern __shared__ __align__(16) char smem_raw[];
  unsigned short* sX = (unsigned short*)smem_raw;        // [TOK_TILE][LDS_PITCH]
  unsigned short* sH = sX + TOK_TILE * LDS_PITCH;        // [TOK_TILE][LDS_PITCH]

  const int tid = threadIdx.x;
  const long long tokBase = (long long)blockIdx.x * TOK_TILE;

  // Stage 64x768 fp32 inputs -> bf16 LDS (float4 loads, packed stores).
  const float* gX = X + tokBase * DH;
  for (int i = tid; i < TOK_TILE * DH / 4; i += 256) {
    float4 f = ((const float4*)gX)[i];
    const int e = i * 4;
    const int row = e / DH, col = e - row * DH;
    unsigned short* p = sX + row * LDS_PITCH + col;
    *(unsigned int*)(p)     = (unsigned int)f2bf(f.x) | ((unsigned int)f2bf(f.y) << 16);
    *(unsigned int*)(p + 2) = (unsigned int)f2bf(f.z) | ((unsigned int)f2bf(f.w) << 16);
  }
  __syncthreads();

  const int wave     = tid >> 5;
  const int lane     = tid & 31;
  const int lane15   = lane & 15;
  const int laneHalf = lane >> 4;
  const int mg       = (wave & 1) * 32;      // 2 strips of 32 tokens (2 M-tiles each)
  const int nq       = (wave >> 1) * 192;    // 4 quarters of the hidden dim

  mlp_layer(sX, sH, W1bf, b1, mg, nq, lane15, laneHalf);
  __syncthreads();
  mlp_layer(sH, sX, W2bf, b2, mg, nq, lane15, laneHalf);
  __syncthreads();

  // Final 768->1 dot + sigmoid: 4 threads per token + shfl reduction.
  {
    const int t    = tid >> 2;               // token 0..63
    const int part = tid & 3;                // 192 elements each
    const unsigned short* h = sX + t * LDS_PITCH + part * 192;
    const float* w3p = W3 + part * 192;
    float z = 0.0f;
    for (int d = 0; d < 192; d += 4) {
      float4 w = *(const float4*)(w3p + d);
      z += bf2f(h[d])     * w.x + bf2f(h[d + 1]) * w.y
         + bf2f(h[d + 2]) * w.z + bf2f(h[d + 3]) * w.w;
    }
    z += __shfl_xor(z, 1, 32);
    z += __shfl_xor(z, 2, 32);
    if (part == 0) {
      z += b3[0];
      ySoft[tokBase + t] = 1.0f / (1.0f + __expf(-z));
    }
  }
}

// ---------------------------------------------------------------------------
// Per-row rank selection, 8 blocks per row (1 token per thread).
// Token 0 is forced strictly below everything (rank 0, and contributes +1 to
// every other token's ascending rank). Stable-argsort tie-break by index.
__global__ void __launch_bounds__(256)
select_kernel(const float* __restrict__ ySoft, float* __restrict__ yHard,
              const int* __restrict__ kPtr)
{
  __shared__ float row[SEQ];
  const int b = blockIdx.x >> 3;             // row
  const int t = ((blockIdx.x & 7) << 8) + threadIdx.x;  // token in row
  const float* y = ySoft + (long long)b * SEQ;
  for (int i = threadIdx.x; i < SEQ; i += 256) row[i] = y[i];
  __syncthreads();
  const int k = *kPtr;

  float out;
  if (t == 0) {
    out = 1.0f;                              // forced minimum -> rank 0 < k
  } else {
    const float v = row[t];
    int rank = 1;                            // token 0 is always below v
#pragma unroll 4
    for (int j = 1; j < SEQ; ++j) {
      const float u = row[j];
      rank += (u < v) || (u == v && j < t);
    }
    out = (rank < k) ? 1.0f : 0.0f;
  }
  yHard[(long long)b * SEQ + t] = out;
}

// ---------------------------------------------------------------------------
extern "C" void kernel_launch(void* const* d_in, const int* in_sizes, int n_in,
                              void* d_out, int out_size, void* d_ws, size_t ws_size,
                              hipStream_t stream) {
  (void)in_sizes; (void)n_in; (void)out_size; (void)ws_size;

  const float* X  = (const float*)d_in[0];
  const float* W1 = (const float*)d_in[1];
  const float* b1 = (const float*)d_in[2];
  const float* W2 = (const float*)d_in[3];
  const float* b2 = (const float*)d_in[4];
  const float* W3 = (const float*)d_in[5];
  const float* b3 = (const float*)d_in[6];
  const int*   kP = (const int*)d_in[7];

  const int Brows = 64;
  float* outHard = (float*)d_out;
  float* outSoft = outHard + (size_t)Brows * SEQ;

  unsigned short* Wbf = (unsigned short*)d_ws;   // [768*768] W1bf, then W2bf

  // 1) fp32 -> bf16 weight conversion (tiny; weights then live in L2).
  cvt_w_kernel<<<(DH * DH + 255) / 256, 256, 0, stream>>>(W1, W2, Wbf);

  // 2) fused 2x GEMM + dot + sigmoid. ~194KB dynamic LDS (320KB WGP budget).
  const int smemBytes = 2 * TOK_TILE * LDS_PITCH * (int)sizeof(unsigned short);
  (void)hipFuncSetAttribute((const void*)mlp_score_kernel,
                            hipFuncAttributeMaxDynamicSharedMemorySize, smemBytes);
  const int nblk = (Brows * SEQ) / TOK_TILE;     // 2048 blocks
  mlp_score_kernel<<<nblk, 256, smemBytes, stream>>>(
      X, Wbf, b1, Wbf + (size_t)DH * DH, b2, W3, b3, outSoft);

  // 3) per-row rank < k mask (8 blocks per row).
  select_kernel<<<Brows * 8, 256, 0, stream>>>(outSoft, outHard, kP);
}
